// KAN_GATv2Layer_15650860826779
// MI455X (gfx1250) — matrix operand order repaired and checked by reading, not compiled, not run on hardware
//
#include <hip/hip_runtime.h>
#include <stdint.h>

typedef __attribute__((ext_vector_type(16))) _Float16 v16h;
typedef __attribute__((ext_vector_type(8)))  _Float16 v8h;
typedef __attribute__((ext_vector_type(8)))  float    v8f;
typedef int v4i __attribute__((vector_size(16)));

namespace {
constexpr int kN  = 50000;   // nodes
constexpr int kE  = 800000;  // edges
constexpr int kF  = 128;     // node features
constexpr int kH  = 4;       // heads
constexpr int kD  = 32;      // dim per head
constexpr int kO  = 128;     // H*D (per direction)
constexpr int kOT = 256;     // src || dst outputs
constexpr int kC  = 11;      // G + K spline coeffs per feature
constexpr int kKA = 1536;    // F + F*kC  (GEMM K dimension)
constexpr int kKT = kKA / 32;     // 48 K-tiles of 32
constexpr int kRT = kN / 16;      // 3125 row tiles (exact)
constexpr float kNegSlope = 0.2f;

// workspace layout (bytes, 256B aligned by construction)
constexpr size_t OFF_A   = 0;
constexpr size_t SZ_A    = (size_t)kN * kKA * 2;        // 153.6 MB f16
constexpr size_t OFF_W   = OFF_A + SZ_A;
constexpr size_t SZ_W    = (size_t)kKA * kOT * 2;       // 786 KB f16 (fragment layout)
constexpr size_t OFF_C   = OFF_W + SZ_W;
constexpr size_t SZ_C    = (size_t)kN * kOT * 4;        // 51.2 MB f32 (sp|dp)
constexpr size_t OFF_ATT = OFF_C + SZ_C;
constexpr size_t SZ_ATT  = (size_t)kE * kH * 4;         // 12.8 MB
constexpr size_t OFF_DEN = OFF_ATT + SZ_ATT;
constexpr size_t SZ_DEN  = (size_t)kN * kH * 4;         // 0.8 MB
constexpr size_t OFF_ACC = OFF_DEN + SZ_DEN;
constexpr size_t SZ_ACC  = (size_t)kN * kF * 4;         // 25.6 MB
constexpr size_t OFF_MAX = OFF_ACC + SZ_ACC;            // 4 bytes
} // namespace

__device__ __forceinline__ float knot(int i) { return -1.75f + 0.25f * (float)i; }

// ---------------------------------------------------------------------------
// gfx1250 async global->LDS copy (ASYNCcnt-tracked); prototype verified by
// round-2 diagnostic: (v4i AS1*, v4i AS3*, i32 offset, i32 cpol).
// ---------------------------------------------------------------------------
__device__ __forceinline__ void async_load_b128(const void* gsrc, void* lds) {
#if defined(__has_builtin) && __has_builtin(__builtin_amdgcn_global_load_async_to_lds_b128)
  __builtin_amdgcn_global_load_async_to_lds_b128(
      (__attribute__((address_space(1))) v4i*)gsrc,
      (__attribute__((address_space(3))) v4i*)lds, 0, 0);
#else
  unsigned ldsoff =
      (unsigned)(uintptr_t)(__attribute__((address_space(3))) void*)lds;
  asm volatile("global_load_async_to_lds_b128 %0, %1, off"
               :
               : "v"(ldsoff), "v"((unsigned long long)(uintptr_t)gsrc)
               : "memory");
#endif
}

__device__ __forceinline__ void wait_asynccnt0() {
#if defined(__has_builtin) && __has_builtin(__builtin_amdgcn_s_wait_asynccnt)
  __builtin_amdgcn_s_wait_asynccnt(0);
#else
  asm volatile("s_wait_asynccnt 0x0" ::: "memory");
#endif
}

__device__ __forceinline__ void sched_fence() {
#if defined(__has_builtin) && __has_builtin(__builtin_amdgcn_sched_barrier)
  __builtin_amdgcn_sched_barrier(0);   // keep B-fragment liveness short
#endif
}

// ---------------------------------------------------------------------------
// A[n, 0:128]        = silu(x[n, f])
// A[n, 128 + f*11+c] = B-spline basis c of x[n, f]  (order 3, uniform grid)
// ---------------------------------------------------------------------------
__global__ void build_A(const float* __restrict__ X, _Float16* __restrict__ A) {
  int idx = blockIdx.x * blockDim.x + threadIdx.x;
  if (idx >= kN * kF) return;
  int n = idx / kF, f = idx % kF;
  float x = X[idx];
  float silu = x / (1.0f + __expf(-x));

  float b[14];
#pragma unroll
  for (int i = 0; i < 14; ++i)
    b[i] = (x >= knot(i) && x < knot(i + 1)) ? 1.0f : 0.0f;
#pragma unroll
  for (int j = 1; j <= 3; ++j) {
    float inv = 1.0f / (0.25f * (float)j);   // uniform knots: t[i+j]-t[i] = j*h
#pragma unroll
    for (int i = 0; i < 14 - j; ++i)
      b[i] = (x - knot(i)) * inv * b[i] + (knot(i + j + 1) - x) * inv * b[i + 1];
  }

  _Float16* row = A + (size_t)n * kKA;
  row[f] = (_Float16)silu;
#pragma unroll
  for (int c = 0; c < kC; ++c)
    row[kF + f * kC + c] = (_Float16)b[c];
}

// ---------------------------------------------------------------------------
// Pack W[1536, 256] (cols 0:128 = src KAN, 128:256 = dst KAN) into WMMA
// B-fragment order: frag (kt, nt): lane = (khalf<<4)|n, element j -> K = khalf*8+j (j<8), 16+khalf*8+(j-8) (j>=8)
// ---------------------------------------------------------------------------
__global__ void pack_W(const float* __restrict__ bsrc, const float* __restrict__ ssrc,
                       const float* __restrict__ bdst, const float* __restrict__ sdst,
                       _Float16* __restrict__ W) {
  int idx = blockIdx.x * blockDim.x + threadIdx.x;   // (kt*16 + nt)*32 + lane
  if (idx >= kKT * 16 * 32) return;
  int lane = idx & 31;
  int nt   = (idx >> 5) & 15;
  int kt   = idx >> 9;
  int ncol  = nt * 16 + (lane & 15);
  int kbase = kt * 32 + (lane >> 4) * 8;
  const float* bw = (ncol < kO) ? bsrc : bdst;
  const float* sw = (ncol < kO) ? ssrc : sdst;
  int o = ncol & (kO - 1);
  _Float16* dst = W + (size_t)idx * 16;
#pragma unroll
  for (int j = 0; j < 16; ++j) {
    int k = kbase + (j < 8 ? j : j + 8);
    float w;
    if (k < kF) {
      w = bw[o * kF + k];
    } else {
      int kk = k - kF;
      w = sw[(size_t)o * kF * kC + (kk / kC) * kC + (kk % kC)];
    }
    dst[j] = (_Float16)w;
  }
}

// ---------------------------------------------------------------------------
// C[N,256] = A[N,1536] x W[1536,256], f16 WMMA, f32 accumulate.
// Block = 256 threads (8 waves); wave w -> row tile blockIdx.x*8+w,
// blockIdx.y -> column half (8 N-tiles = 128 cols per wave).
// Per-k-tile 8 KB B panel shared by all 8 waves: staged in LDS with async
// global->LDS copies, double-buffered via runtime pointer ping-pong
// (#pragma unroll 1 keeps one loop body so accumulators stay pinned in the
// WMMA D/C registers across the backedge — no cross-copy v_movs).
// ---------------------------------------------------------------------------
__global__ void __launch_bounds__(256)
kan_gemm(const _Float16* __restrict__ A, const _Float16* __restrict__ W,
         float* __restrict__ C) {
  __shared__ _Float16 Bs[2][8 * 32 * 16];   // double-buffered 8 KB panels

  const int tid  = threadIdx.x;
  const int wave = tid >> 5;
  const int lane = tid & 31;
  const int rowTile = blockIdx.x * 8 + wave;
  const bool active = rowTile < kRT;
  const int rowEff  = active ? rowTile : (kRT - 1);
  const int ntb = blockIdx.y * 8;           // first of 8 column tiles

  const int m     = lane & 15;
  const int koff8 = (lane >> 4) * 8;        // A layout: lane-half selects K sub-chunk
  const _Float16* arow = A + (size_t)(rowEff * 16 + m) * kKA + koff8;

  // this block's contiguous 8-fragment column panel (per k-tile: +kt*8192 f16)
  const _Float16* wpanel = W + (size_t)ntb * (32 * 16);

  v8f acc[8];
#pragma unroll
  for (int t = 0; t < 8; ++t) acc[t] = (v8f){};

  // runtime ping-pong pointers (keeps the k-loop a single body)
  _Float16* bcur = &Bs[0][0];
  _Float16* bnxt = &Bs[1][0];

  // prologue: stage k-tile 0 (each thread copies 32 B of the 8 KB panel)
  {
    const _Float16* g = wpanel + (size_t)tid * 16;
    async_load_b128(g,     bcur + tid * 16);
    async_load_b128(g + 8, bcur + tid * 16 + 8);
  }
  // software-pipelined A fragment for k-tile 0
  v8h lo = *(const v8h*)(arow);
  v8h hi = *(const v8h*)(arow + 16);
  wait_asynccnt0();
  __syncthreads();

#pragma unroll 1
  for (int kt = 0; kt < kKT; ++kt) {
    if (kt + 1 < kKT) {                     // stage next panel into other buffer
      const _Float16* g =
          wpanel + (size_t)(kt + 1) * (16 * 32 * 16) + (size_t)tid * 16;
      async_load_b128(g,     bnxt + tid * 16);
      async_load_b128(g + 8, bnxt + tid * 16 + 8);
    }

    // prefetch + preload A fragment for k-tile kt+1
    v8h lo_n = lo, hi_n = hi;
    if (kt + 1 < kKT) {
      __builtin_prefetch(arow + (kt + 1) * 32, 0, 0);   // global_prefetch_b8
      lo_n = *(const v8h*)(arow + (kt + 1) * 32);
      hi_n = *(const v8h*)(arow + (kt + 1) * 32 + 16);
    }

    // A fragment: K = {koff8..koff8+7} in v0..3, {16+koff8..} in v4..7
    v16h a = __builtin_shufflevector(lo, hi, 0, 1, 2, 3, 4, 5, 6, 7,
                                             8, 9, 10, 11, 12, 13, 14, 15);

#pragma unroll
    for (int t = 0; t < 4; ++t) {
      v16h b = *(const v16h*)(bcur + (t * 32 + lane) * 16);
      acc[t] = __builtin_amdgcn_wmma_f32_16x16x32_f16(
          false, a, false, b, (short)0, acc[t], false, false);
    }
    sched_fence();                          // limit live B fragments to 4
#pragma unroll
    for (int t = 4; t < 8; ++t) {
      v16h b = *(const v16h*)(bcur + (t * 32 + lane) * 16);
      acc[t] = __builtin_amdgcn_wmma_f32_16x16x32_f16(
          false, a, false, b, (short)0, acc[t], false, false);
    }

    // swap buffers, advance pipelined A fragment
    _Float16* tswap = bcur; bcur = bnxt; bnxt = tswap;
    lo = lo_n;
    hi = hi_n;
    wait_asynccnt0();                       // next panel landed in LDS
    __syncthreads();                        // all waves done reading old panel
  }

  if (active) {
    int row0 = rowTile * 16;
    int mo   = (lane >> 4) * 8;             // C layout: lanes 16-31 hold M=8..15
    int nlo  = lane & 15;
#pragma unroll
    for (int r = 0; r < 8; ++r) {
      float* crow = C + (size_t)(row0 + mo + r) * kOT;
#pragma unroll
      for (int t = 0; t < 8; ++t)
        crow[(ntb + t) * 16 + nlo] = acc[t][r];
    }
  }
}

// ---------------------------------------------------------------------------
// Edge scores: attn[e,h] = sum_d a[h,d]*leakyrelu(sp[src,h,d]+dp[dst,h,d]);
// global max via monotone-uint atomicMax (block-reduced first).
// ---------------------------------------------------------------------------
__global__ void edge_attn(const int* __restrict__ ei, const float* __restrict__ C,
                          const float* __restrict__ attw, float* __restrict__ attn,
                          unsigned* __restrict__ maxbuf) {
  int idx = blockIdx.x * blockDim.x + threadIdx.x;
  float val = -3.402823466e38f;
  if (idx < kE * kH) {
    int e = idx >> 2, h = idx & 3;
    int src = ei[e], dst = ei[kE + e];
    const float* sp = C + (size_t)src * kOT + h * kD;
    const float* dp = C + (size_t)dst * kOT + kO + h * kD;
    const float* aw = attw + h * kD;
    float acc = 0.0f;
#pragma unroll
    for (int d = 0; d < kD; ++d) {
      float v = sp[d] + dp[d];
      v = (v >= 0.0f) ? v : kNegSlope * v;
      acc = fmaf(aw[d], v, acc);
    }
    attn[idx] = acc;
    val = acc;
  }
  __shared__ float smax[256];
  smax[threadIdx.x] = val;
  __syncthreads();
  for (int s = 128; s > 0; s >>= 1) {
    if (threadIdx.x < s) smax[threadIdx.x] = fmaxf(smax[threadIdx.x], smax[threadIdx.x + s]);
    __syncthreads();
  }
  if (threadIdx.x == 0) {
    unsigned b = __float_as_uint(smax[0]);
    unsigned enc = (smax[0] >= 0.0f) ? (b | 0x80000000u) : ~b;
    atomicMax(maxbuf, enc);
  }
}

__global__ void edge_exp(const int* __restrict__ ei, float* __restrict__ attn,
                         float* __restrict__ denom, const unsigned* __restrict__ maxbuf) {
  int idx = blockIdx.x * blockDim.x + threadIdx.x;
  if (idx >= kE * kH) return;
  unsigned u = *maxbuf;
  float mx = (u & 0x80000000u) ? __uint_as_float(u ^ 0x80000000u) : __uint_as_float(~u);
  int e = idx >> 2, h = idx & 3;
  float v = __expf(attn[idx] - mx);
  attn[idx] = v;                                // in-place: exp_attn
  atomicAdd(denom + (size_t)ei[kE + e] * kH + h, v);
}

// accum[dst, r] += alpha(e, r/32) * sp[src, r]
__global__ void edge_msg(const int* __restrict__ ei, const float* __restrict__ C,
                         const float* __restrict__ exp_attn,
                         const float* __restrict__ denom, float* __restrict__ accum) {
  int idx = blockIdx.x * blockDim.x + threadIdx.x;
  if (idx >= kE * kF) return;
  int e = idx >> 7, r = idx & 127, h = r >> 5;
  int src = ei[e], dst = ei[kE + e];
  float alpha = exp_attn[(size_t)e * kH + h] /
                (denom[(size_t)dst * kH + h] + 1e-16f);
  atomicAdd(accum + (size_t)dst * kF + r, alpha * C[(size_t)src * kOT + r]);
}

__global__ void finalize(const float* __restrict__ accum, const float* __restrict__ X,
                         const float* __restrict__ bias, const float* __restrict__ pa,
                         float* __restrict__ out) {
  int idx = blockIdx.x * blockDim.x + threadIdx.x;
  if (idx >= kN * kF) return;
  float v = accum[idx] + X[idx] + bias[idx & 127];
  out[idx] = (v >= 0.0f) ? v : pa[0] * v;
}

// ---------------------------------------------------------------------------
extern "C" void kernel_launch(void* const* d_in, const int* in_sizes, int n_in,
                              void* d_out, int out_size, void* d_ws, size_t ws_size,
                              hipStream_t stream) {
  const float* x     = (const float*)d_in[0];
  const int*   ei    = (const int*)d_in[1];
  const float* bsrc  = (const float*)d_in[2];
  const float* ssrc  = (const float*)d_in[3];
  const float* bdst  = (const float*)d_in[4];
  const float* sdst  = (const float*)d_in[5];
  const float* attw  = (const float*)d_in[6];
  const float* bias  = (const float*)d_in[7];
  const float* pa    = (const float*)d_in[8];

  char* ws = (char*)d_ws;
  _Float16* A     = (_Float16*)(ws + OFF_A);
  _Float16* Wp    = (_Float16*)(ws + OFF_W);
  float*    Cm    = (float*)(ws + OFF_C);
  float*    attn  = (float*)(ws + OFF_ATT);
  float*    denom = (float*)(ws + OFF_DEN);
  float*    accum = (float*)(ws + OFF_ACC);
  unsigned* mxb   = (unsigned*)(ws + OFF_MAX);

  (void)hipMemsetAsync(denom, 0, SZ_DEN, stream);
  (void)hipMemsetAsync(accum, 0, SZ_ACC, stream);
  (void)hipMemsetAsync(mxb,   0, 4, stream);

  build_A<<<(kN * kF + 255) / 256, 256, 0, stream>>>(x, A);
  pack_W<<<(kKT * 16 * 32 + 255) / 256, 256, 0, stream>>>(bsrc, ssrc, bdst, sdst, Wp);

  dim3 ggrid((kRT + 7) / 8, 2);
  kan_gemm<<<ggrid, 256, 0, stream>>>(A, Wp, Cm);

  edge_attn<<<(kE * kH + 255) / 256, 256, 0, stream>>>(ei, Cm, attw, attn, mxb);
  edge_exp<<<(kE * kH + 255) / 256, 256, 0, stream>>>(ei, attn, denom, mxb);
  edge_msg<<<(kE * kF + 255) / 256, 256, 0, stream>>>(ei, Cm, attn, denom, accum);
  finalize<<<(kN * kF + 255) / 256, 256, 0, stream>>>(accum, x, bias, pa, (float*)d_out);
}
